// TransGNN_89524298318365
// MI455X (gfx1250) — matrix operand order repaired and checked by reading, not compiled, not run on hardware
//
#include <hip/hip_runtime.h>

// ---------------------------------------------------------------------------
// CDNA5 (gfx1250) TransformerConv GNN.
// Dense projections: v_wmma_f32_16x16x32_bf16, B staged to LDS via async copy,
// B fragments hoisted to registers, 2 column tiles per wave, K fully unrolled.
// Edge attention: wave-per-edge VALU + ordered-int atomic max + float atomics.
// ---------------------------------------------------------------------------

typedef __attribute__((ext_vector_type(16))) __bf16          v16bf;
typedef __attribute__((ext_vector_type(16))) unsigned short  v16u;
typedef __attribute__((ext_vector_type(8)))  float           v8f;
typedef __attribute__((ext_vector_type(4)))  int             v4i_t;

#define N_NODES 50000
#define N_EDGES 400000
#define HD      128    // HEADS * OUT_CH
#define HEADS   4
#define OC      32
#define EDIM    16

// ---- helpers ---------------------------------------------------------------
__device__ __forceinline__ unsigned short f32_to_bf16(float f) {
    unsigned u = __float_as_uint(f);
    unsigned r = 0x7FFFu + ((u >> 16) & 1u);   // round-to-nearest-even
    return (unsigned short)((u + r) >> 16);
}
__device__ __forceinline__ unsigned ford(float f) {            // order-preserving map
    int i = __float_as_int(f);
    unsigned u = (unsigned)i;
    return (i >= 0) ? (u | 0x80000000u) : ~u;
}
__device__ __forceinline__ float funord(unsigned u) {
    int i = (u & 0x80000000u) ? (int)(u & 0x7FFFFFFFu) : (int)~u;
    return __int_as_float(i);
}

// ---- elementwise utility kernels ------------------------------------------
__global__ void k_cvt_bf16(const float* __restrict__ s, unsigned short* __restrict__ d, int n) {
    int i = blockIdx.x * blockDim.x + threadIdx.x;
    if (i < n) d[i] = f32_to_bf16(s[i]);
}
__global__ void k_zero_f32(float* __restrict__ p, int n) {
    int i = blockIdx.x * blockDim.x + threadIdx.x;
    if (i < n) p[i] = 0.0f;
}
__global__ void k_zero_u32(unsigned* __restrict__ p, int n) {
    int i = blockIdx.x * blockDim.x + threadIdx.x;
    if (i < n) p[i] = 0u;
}

// ---- WMMA GEMM: C[M,Ncols] = A[M,K](bf16) * B[K,Ncols](bf16) + bias --------
// Requirements: M = Mtiles*16 exactly, Ncols multiple of 32, K in {32,128}.
// Each wave owns a 32-column group; B fragments live in registers, sourced
// from an LDS copy of B staged with the gfx1250 async-to-LDS path.
template<int K>
__global__ void k_gemm_bf16(const unsigned short* __restrict__ A,
                            const unsigned short* __restrict__ Bg,
                            const float* __restrict__ bias,
                            float* __restrict__ C,
                            int Mtiles, int Ncols) {
    constexpr int KS = K / 32;
    __shared__ unsigned short ldsB[16384];          // up to 128x128 bf16 = 32KB

    // ---- stage B (K x Ncols) into LDS ----
    const int tid = threadIdx.x;
    const int nv = (K * Ncols) / 8;                 // 16-byte chunks
#if __has_builtin(__builtin_amdgcn_global_load_async_to_lds_b128) && \
    __has_builtin(__builtin_amdgcn_s_wait_asynccnt)
    {
        auto gsrc = (__attribute__((address_space(1))) char*)Bg;
        auto ldst = (__attribute__((address_space(3))) char*)ldsB;
        for (int i = tid; i < nv; i += blockDim.x)
            __builtin_amdgcn_global_load_async_to_lds_b128(
                (__attribute__((address_space(1))) v4i_t*)(gsrc + (size_t)i * 16),
                (__attribute__((address_space(3))) v4i_t*)(ldst + (size_t)i * 16),
                0, 0);
        __builtin_amdgcn_s_wait_asynccnt(0);
    }
#else
    for (int i = tid; i < nv; i += blockDim.x)
        ((v4i_t*)ldsB)[i] = ((const v4i_t*)Bg)[i];
#endif
    __syncthreads();

    const int lane = tid & 31;
    const int half = lane >> 4;                     // 0: lanes 0-15, 1: lanes 16-31
    const int l16  = lane & 15;
    const int G    = Ncols >> 5;                    // 32-column groups (2 tiles each)
    const int wv   = (blockIdx.x * blockDim.x + tid) >> 5;
    const int nW   = (gridDim.x * blockDim.x) >> 5;
    const int wpg  = nW / G;                        // waves per column group
    const int g    = wv / wpg;
    if (g >= G) return;
    const int mt0  = wv - g * wpg;
    const int col0 = g * 32 + l16;
    const int col1 = col0 + 16;

    // ---- hoist B fragments (ISA 7.12.2 B 32x16 layout) from LDS ----
    v16u bu0[KS], bu1[KS];
#pragma unroll
    for (int ks = 0; ks < KS; ++ks) {
        const unsigned short* bp = ldsB + (ks * 32 + half * 8) * Ncols;
#pragma unroll
        for (int e = 0; e < 16; ++e) {
            const int kr = (e < 8) ? e : e + 8;
            bu0[ks][e] = bp[kr * Ncols + col0];
            bu1[ks][e] = bp[kr * Ncols + col1];
        }
    }
    const float bias0 = bias ? bias[col0] : 0.0f;
    const float bias1 = bias ? bias[col1] : 0.0f;

    for (int mt = mt0; mt < Mtiles; mt += wpg) {
        const unsigned short* arow = A + (size_t)(mt * 16 + l16) * K;
        __builtin_prefetch(A + (size_t)((mt + wpg) * 16 + l16) * K, 0, 1);
        v8f acc0 = {}, acc1 = {};
#pragma unroll
        for (int ks = 0; ks < KS; ++ks) {           // A 16x32 bf16 layout
            v16u au;
            const unsigned short* ap = arow + ks * 32 + half * 8;
#pragma unroll
            for (int e = 0; e < 8; ++e) au[e]     = ap[e];        // K = base+0..7
#pragma unroll
            for (int e = 0; e < 8; ++e) au[8 + e] = ap[16 + e];   // K = base+16..23
            v16bf a = __builtin_bit_cast(v16bf, au);
            acc0 = __builtin_amdgcn_wmma_f32_16x16x32_bf16(
                     false, a, false, __builtin_bit_cast(v16bf, bu0[ks]),
                     (short)0, acc0, false, false);
            acc1 = __builtin_amdgcn_wmma_f32_16x16x32_bf16(
                     false, a, false, __builtin_bit_cast(v16bf, bu1[ks]),
                     (short)0, acc1, false, false);
        }
        // D: row = mt*16 + r + 8*half, cols col0/col1
        float* crow = C + (size_t)(mt * 16 + 8 * half) * Ncols;
#pragma unroll
        for (int r = 0; r < 8; ++r) {
            crow[(size_t)r * Ncols + col0] = acc0[r] + bias0;
            crow[(size_t)r * Ncols + col1] = acc1[r] + bias1;
        }
    }
}

// ---- edge phase ------------------------------------------------------------
// wave per edge; lane covers 4 contiguous channels (head = lane>>3)
__global__ void k_edge_alpha(const int* __restrict__ ei, const float* __restrict__ ea,
                             const float* __restrict__ We, const float* __restrict__ q,
                             const float* __restrict__ kk, float* __restrict__ alpha,
                             unsigned* __restrict__ amax, int E) {
    int wave   = (blockIdx.x * blockDim.x + threadIdx.x) >> 5;
    int lane   = threadIdx.x & 31;
    int stride = (gridDim.x * blockDim.x) >> 5;
    const float scale = 0.17677669529663687f;   // 1/sqrt(32)
    for (int e = wave; e < E; e += stride) {
        int s = ei[e], d = ei[E + e];
        int c0 = lane * 4;
        float ear[EDIM];
#pragma unroll
        for (int j = 0; j < EDIM; ++j) ear[j] = ea[(size_t)e * EDIM + j];
        float ev[4];
#pragma unroll
        for (int i = 0; i < 4; ++i) {
            float acc = 0.f;
#pragma unroll
            for (int j = 0; j < EDIM; ++j) acc += ear[j] * We[j * HD + c0 + i];
            ev[i] = acc;
        }
        const float* qp = q  + (size_t)d * HD + c0;
        const float* kp = kk + (size_t)s * HD + c0;
        float p = 0.f;
#pragma unroll
        for (int i = 0; i < 4; ++i) p += qp[i] * (kp[i] + ev[i]);
        p += __shfl_xor(p, 1, 32);   // reduce 8 lanes of one head
        p += __shfl_xor(p, 2, 32);
        p += __shfl_xor(p, 4, 32);
        if ((lane & 7) == 0) {
            int h = lane >> 3;
            float a = p * scale;
            alpha[(size_t)e * HEADS + h] = a;
            atomicMax(&amax[d * HEADS + h], ford(a));
        }
    }
}

__global__ void k_edge_exp(const int* __restrict__ ei, float* __restrict__ alpha,
                           const unsigned* __restrict__ amax, float* __restrict__ denom, int E) {
    int i = blockIdx.x * blockDim.x + threadIdx.x;
    if (i >= E * HEADS) return;
    int e = i >> 2, h = i & 3;
    int d = ei[E + e];
    float a = __expf(alpha[i] - funord(amax[d * HEADS + h]));
    alpha[i] = a;
    atomicAdd(&denom[d * HEADS + h], a);
}

__global__ void k_edge_scatter(const int* __restrict__ ei, const float* __restrict__ ea,
                               const float* __restrict__ We, const float* __restrict__ vv,
                               const float* __restrict__ alpha, const float* __restrict__ denom,
                               float* __restrict__ accum, int E) {
    int wave   = (blockIdx.x * blockDim.x + threadIdx.x) >> 5;
    int lane   = threadIdx.x & 31;
    int stride = (gridDim.x * blockDim.x) >> 5;
    for (int e = wave; e < E; e += stride) {
        int s = ei[e], d = ei[E + e];
        int c0 = lane * 4, h = lane >> 3;
        float w = alpha[(size_t)e * HEADS + h] / (denom[d * HEADS + h] + 1e-16f);
        float ear[EDIM];
#pragma unroll
        for (int j = 0; j < EDIM; ++j) ear[j] = ea[(size_t)e * EDIM + j];
#pragma unroll
        for (int i = 0; i < 4; ++i) {
            float ev = 0.f;
#pragma unroll
            for (int j = 0; j < EDIM; ++j) ev += ear[j] * We[j * HD + c0 + i];
            float val = (vv[(size_t)s * HD + c0 + i] + ev) * w;
            atomicAdd(&accum[(size_t)d * HD + c0 + i], val);
        }
    }
}

// ---- combine + ReLU --------------------------------------------------------
__global__ void k_combine_concat(const float* __restrict__ accum, const float* __restrict__ skip,
                                 unsigned short* __restrict__ hbf, int n) {
    int i = blockIdx.x * blockDim.x + threadIdx.x;
    if (i < n) hbf[i] = f32_to_bf16(fmaxf(accum[i] + skip[i], 0.0f));
}
__global__ void k_combine_mean(const float* __restrict__ accum, const float* __restrict__ skip,
                               unsigned short* __restrict__ h3, int n /* N*OC */) {
    int i = blockIdx.x * blockDim.x + threadIdx.x;
    if (i >= n) return;
    int nd = i >> 5, c = i & 31;
    const float* ap = accum + (size_t)nd * HD + c;
    float m = 0.25f * (ap[0] + ap[OC] + ap[2 * OC] + ap[3 * OC]);
    h3[i] = f32_to_bf16(fmaxf(m + skip[i], 0.0f));
}

// ---------------------------------------------------------------------------
extern "C" void kernel_launch(void* const* d_in, const int* in_sizes, int n_in,
                              void* d_out, int out_size, void* d_ws, size_t ws_size,
                              hipStream_t stream) {
    const int N = N_NODES, E = N_EDGES;
    auto cdiv = [](long long a, long long b) { return (int)((a + b - 1) / b); };

    // --- workspace carve-out (256B aligned) ---
    char* wsp = (char*)d_ws;
    auto alloc = [&](size_t bytes) { char* p = wsp; wsp += (bytes + 255) & ~(size_t)255; return p; };
    unsigned short* h_bf  = (unsigned short*)alloc((size_t)N * HD * 2);
    float*          qbuf  = (float*)alloc((size_t)N * HD * 4);
    float*          kbuf  = (float*)alloc((size_t)N * HD * 4);
    float*          vbuf  = (float*)alloc((size_t)N * HD * 4);
    float*          sbuf  = (float*)alloc((size_t)N * HD * 4);
    float*          accum = (float*)alloc((size_t)N * HD * 4);
    float*          alpha = (float*)alloc((size_t)E * HEADS * 4);
    unsigned*       amax  = (unsigned*)alloc((size_t)N * HEADS * 4);
    float*          denom = (float*)alloc((size_t)N * HEADS * 4);
    unsigned short* wq_bf = (unsigned short*)alloc((size_t)HD * HD * 2);
    unsigned short* wk_bf = (unsigned short*)alloc((size_t)HD * HD * 2);
    unsigned short* wv_bf = (unsigned short*)alloc((size_t)HD * HD * 2);
    unsigned short* ws_bf = (unsigned short*)alloc((size_t)HD * HD * 2);
    unsigned short* wf_bf = (unsigned short*)alloc((size_t)OC * OC * 2);
    unsigned short* h3_bf = (unsigned short*)alloc((size_t)N * OC * 2);

    const float* x  = (const float*)d_in[0];
    const int*   ei = (const int*)d_in[1];      // [2,E]: src row then dst row
    const float* ea = (const float*)d_in[2];

    const int TB = 256;
    k_cvt_bf16<<<cdiv((long long)N * HD, TB), TB, 0, stream>>>(x, h_bf, N * HD);

    const int Mtiles = N / 16;                  // 3125, exact
    const int edgeBlocks = cdiv((long long)E * 32, TB);
    const int GB = 512;                         // GEMM blocks (4096 waves)

    for (int l = 0; l < 3; ++l) {
        int base = 3 + l * 9;
        const float* Wq = (const float*)d_in[base + 0];
        const float* bq = (const float*)d_in[base + 1];
        const float* Wk = (const float*)d_in[base + 2];
        const float* bk = (const float*)d_in[base + 3];
        const float* Wv = (const float*)d_in[base + 4];
        const float* bv = (const float*)d_in[base + 5];
        const float* We = (const float*)d_in[base + 6];
        const float* Ws = (const float*)d_in[base + 7];
        const float* bs = (const float*)d_in[base + 8];
        const bool concat = (l < 2);
        const int skipN = concat ? HD : OC;

        // weights -> bf16
        k_cvt_bf16<<<cdiv(HD * HD, TB), TB, 0, stream>>>(Wq, wq_bf, HD * HD);
        k_cvt_bf16<<<cdiv(HD * HD, TB), TB, 0, stream>>>(Wk, wk_bf, HD * HD);
        k_cvt_bf16<<<cdiv(HD * HD, TB), TB, 0, stream>>>(Wv, wv_bf, HD * HD);
        k_cvt_bf16<<<cdiv(HD * skipN, TB), TB, 0, stream>>>(Ws, ws_bf, HD * skipN);

        // projections via WMMA (K = 128)
        k_gemm_bf16<128><<<GB, TB, 0, stream>>>(h_bf, wq_bf, bq, qbuf, Mtiles, HD);
        k_gemm_bf16<128><<<GB, TB, 0, stream>>>(h_bf, wk_bf, bk, kbuf, Mtiles, HD);
        k_gemm_bf16<128><<<GB, TB, 0, stream>>>(h_bf, wv_bf, bv, vbuf, Mtiles, HD);
        k_gemm_bf16<128><<<GB, TB, 0, stream>>>(h_bf, ws_bf, bs, sbuf, Mtiles, skipN);

        // reset segment state
        k_zero_u32<<<cdiv(N * HEADS, TB), TB, 0, stream>>>(amax, N * HEADS);
        k_zero_f32<<<cdiv(N * HEADS, TB), TB, 0, stream>>>(denom, N * HEADS);
        k_zero_f32<<<cdiv((long long)N * HD, TB), TB, 0, stream>>>(accum, N * HD);

        // attention
        k_edge_alpha<<<edgeBlocks, TB, 0, stream>>>(ei, ea, We, qbuf, kbuf, alpha, amax, E);
        k_edge_exp<<<cdiv((long long)E * HEADS, TB), TB, 0, stream>>>(ei, alpha, amax, denom, E);
        k_edge_scatter<<<edgeBlocks, TB, 0, stream>>>(ei, ea, We, vbuf, alpha, denom, accum, E);

        // combine + ReLU
        if (concat) {
            k_combine_concat<<<cdiv((long long)N * HD, TB), TB, 0, stream>>>(accum, sbuf, h_bf, N * HD);
        } else {
            k_combine_mean<<<cdiv((long long)N * OC, TB), TB, 0, stream>>>(accum, sbuf, h3_bf, N * OC);
        }
    }

    // final linear: out = h3 @ Wf + bf   (K = 32)
    const float* Wf = (const float*)d_in[30];
    const float* bf = (const float*)d_in[31];
    k_cvt_bf16<<<cdiv(OC * OC, TB), TB, 0, stream>>>(Wf, wf_bf, OC * OC);
    k_gemm_bf16<32><<<GB, TB, 0, stream>>>(h3_bf, wf_bf, bf, (float*)d_out, Mtiles, OC);
}
